// VGG19HebDepreciated_3685081940680
// MI455X (gfx1250) — matrix operand order repaired and checked by reading, not compiled, not run on hardware
//
#include <hip/hip_runtime.h>

typedef __attribute__((ext_vector_type(16))) _Float16 v16h;
typedef __attribute__((ext_vector_type(8)))  float    v8f;

// ---------------------------------------------------------------------------
// Kernel 1: s[b] = count of positive elements in curr_x[b, :, :, :]
// 128 blocks (one per batch), 256 threads, LDS tree reduction.
// ---------------------------------------------------------------------------
__global__ __launch_bounds__(256) void count_pos_kernel(
    const float* __restrict__ x, float* __restrict__ s, int per_batch) {
  const int b = blockIdx.x;
  const float* xb = x + (size_t)b * (size_t)per_batch;
  int cnt = 0;
  for (int i = threadIdx.x; i < per_batch; i += blockDim.x)
    cnt += (xb[i] > 0.0f) ? 1 : 0;

  __shared__ int red[256];
  red[threadIdx.x] = cnt;
  __syncthreads();
  for (int off = 128; off > 0; off >>= 1) {
    if ((int)threadIdx.x < off) red[threadIdx.x] += red[threadIdx.x + off];
    __syncthreads();
  }
  if (threadIdx.x == 0) s[b] = (float)red[0];
}

// ---------------------------------------------------------------------------
// Kernel 2: out[n] = sum_b (prev[b*N + n] > 0) * s[b]   via WMMA GEMV.
//
// One wave32 handles TILES_PER_WAVE adjacent 16-output tiles, so the
// A-operand (s digits) gather + f16 conversion is amortized 4x.
// K = 128 batches in 4 chunks of 32.
// A (16x32 f16) = s digits broadcast down rows; B (32x16 f16) = binary tile.
// s[b] = 2048*q + r exact fp16 digit split; out = 2048*Cq + Cr (f32, exact).
// ---------------------------------------------------------------------------
#define TILES_PER_WAVE 4

__global__ __launch_bounds__(256) void heb_gemv_wmma_kernel(
    const float* __restrict__ prev, const float* __restrict__ s,
    float* __restrict__ out, int Ntot, int nGroups) {
  const int lane  = (int)(threadIdx.x & 31u);
  const int wave  = (int)(threadIdx.x >> 5);
  const int group = (int)blockIdx.x * (int)(blockDim.x >> 5) + wave;
  if (group >= nGroups) return;  // uniform per-wave: EXEC all-ones inside

  const int nlane = lane & 15;
  const int hi    = (lane >> 4) & 1;   // which half-wave (K sub-range)

  // --- Build A operands once: fp16 digits of s, in contiguous k order (same as B).
  // lanes 0-15 : elements 0..15 -> K=0..15
  // lanes 16-31: elements 0..15 -> K=16..31
  v16h aq[4], ar[4];
#pragma unroll
  for (int c = 0; c < 4; ++c) {
#pragma unroll
    for (int i = 0; i < 16; ++i) {
      const int klocal = hi * 16 + i;
      const int sv = (int)s[c * 32 + klocal];      // exact integer count
      aq[c][i] = (_Float16)(sv >> 11);             // q <= 49   (exact fp16)
      ar[c][i] = (_Float16)(sv & 2047);            // r <= 2047 (exact fp16)
    }
  }

  // --- Loop over 4 adjacent tiles, reusing the A operands.
#pragma unroll
  for (int t = 0; t < TILES_PER_WAVE; ++t) {
    const int tile = group * TILES_PER_WAVE + t;
    const int n0   = tile * 16;

    v8f cq = {};
    v8f cr = {};
#pragma unroll
    for (int c = 0; c < 4; ++c) {
      // B (32x16): lane = column n0+nlane; element i -> K = hi*16 + i.
      // Half-wave loads 16 consecutive floats per i; the adjacent tile's
      // half-wave completes the other 64B of each line.
      const float* col = prev +
          (size_t)(c * 32 + hi * 16) * (size_t)Ntot + (size_t)(n0 + nlane);
      v16h bmat;
#pragma unroll
      for (int i = 0; i < 16; ++i) {
        const float v = col[(size_t)i * (size_t)Ntot];
        bmat[i] = (v > 0.0f) ? (_Float16)1.0f : (_Float16)0.0f;
      }
      cq = __builtin_amdgcn_wmma_f32_16x16x32_f16(
          /*neg_a=*/false, aq[c], /*neg_b=*/false, bmat,
          /*c_mod=*/(short)0, cq, /*reuse_a=*/false, /*reuse_b=*/false);
      cr = __builtin_amdgcn_wmma_f32_16x16x32_f16(
          /*neg_a=*/false, ar[c], /*neg_b=*/false, bmat,
          /*c_mod=*/(short)0, cr, /*reuse_a=*/false, /*reuse_b=*/false);
    }

    // D layout: VGPR0, lanes 0-15 = (M=0, N=lane). All rows identical by
    // construction; row 0 carries the 16 outputs.
    if (lane < 16) out[n0 + lane] = 2048.0f * cq[0] + cr[0];
  }
}

// ---------------------------------------------------------------------------
extern "C" void kernel_launch(void* const* d_in, const int* in_sizes, int n_in,
                              void* d_out, int out_size, void* d_ws, size_t ws_size,
                              hipStream_t stream) {
  const float* prev_x = (const float*)d_in[0];  // [128, 256, 28, 28] f32
  const float* curr_x = (const float*)d_in[1];  // [128, 512, 14, 14] f32
  float* out = (float*)d_out;                   // [256, 28, 28] f32
  float* s   = (float*)d_ws;                    // 128 floats scratch

  const int B        = 128;
  const int Ntot     = in_sizes[0] / B;         // 200704
  const int curr_per = in_sizes[1] / B;         // 100352
  const int nTiles   = Ntot / 16;               // 12544
  const int nGroups  = (nTiles + TILES_PER_WAVE - 1) / TILES_PER_WAVE;  // 3136

  // Phase 1: per-batch positive counts -> d_ws.
  count_pos_kernel<<<B, 256, 0, stream>>>(curr_x, s, curr_per);

  // Phase 2: WMMA GEMV, one wave per 4-tile group (8 waves / 256-thr block).
  const int wavesPerBlock = 256 / 32;
  const int blocks = (nGroups + wavesPerBlock - 1) / wavesPerBlock;  // 392
  heb_gemv_wmma_kernel<<<blocks, 256, 0, stream>>>(prev_x, s, out, Ntot, nGroups);
}
